// SparseRoformerLayer_56160992362953
// MI455X (gfx1250) — matrix operand order repaired and checked by reading, not compiled, not run on hardware
//
#include <hip/hip_runtime.h>
#include <hip/hip_bf16.h>
#include <math.h>

// Problem constants
#define S_   2048
#define B_   4
#define E_   1024
#define F_   4096
#define H_   16
#define NE_  8
#define CAP_ 2048
#define N_   (S_ * B_)   // 8192 tokens
#define HD_  64          // head dim

typedef __bf16 bf16;
typedef __attribute__((ext_vector_type(16))) __bf16 v16bf;
typedef __attribute__((ext_vector_type(8)))  float  v8f;

// ---------------------------------------------------------------- helpers ---

__device__ __forceinline__ v8f vzero8() {
  v8f z;
#pragma unroll
  for (int i = 0; i < 8; ++i) z[i] = 0.0f;
  return z;
}

// Async global -> LDS copy of 16 bytes per lane (CDNA5 GLOBAL_LOAD_ASYNC_TO_LDS_B128,
// tracked by ASYNCcnt). lds_off is the byte address within the workgroup's LDS
// allocation; gptr is the per-lane global address.
__device__ __forceinline__ void async_load16(unsigned int lds_off, const void* gptr) {
  asm volatile("global_load_async_to_lds_b128 %0, %1, off"
               :: "v"(lds_off), "v"((unsigned long long)(uintptr_t)gptr)
               : "memory");
}

__device__ __forceinline__ void wait_async0() {
  asm volatile("s_wait_asynccnt 0x0" ::: "memory");
}
// Allow 6 async ops (one K-tile worth) to remain outstanding: the older tile
// is complete, the newer one keeps flying under the WMMAs.
__device__ __forceinline__ void wait_async6() {
  asm volatile("s_wait_asynccnt 0x6" ::: "memory");
}

// A fragment (16x32 bf16), row-major source with leading dim `ld` (elems).
// ISA layout: lanes 0-15 -> row = lane, K = {0..7, 16..23};
//             lanes 16-31 -> row = lane-16, K = {8..15, 24..31}.
__device__ __forceinline__ v16bf load_a_frag(const bf16* base, int ld) {
  int lane = threadIdx.x & 31;
  int row  = lane & 15;
  int koff = (lane >> 4) << 3;           // 0 or 8
  const bf16* p = base + row * ld + koff;
  v16bf r;
  *reinterpret_cast<uint4*>(&r)       = *reinterpret_cast<const uint4*>(p);
  *(reinterpret_cast<uint4*>(&r) + 1) = *reinterpret_cast<const uint4*>(p + 16);
  return r;
}

// B fragment (32x16 bf16) from transposed storage Bt[n][k] (n-major,
// leading dim `ld`): lane -> N = lane&15, K = (lane>>4)*8 + {0..7, 16..23}.
__device__ __forceinline__ v16bf load_b_frag(const bf16* baseT, int ld) {
  int lane = threadIdx.x & 31;
  int n    = lane & 15;
  int koff = (lane >> 4) << 3;           // 0 or 8
  const bf16* p = baseT + n * ld + koff;
  v16bf r;
  *reinterpret_cast<uint4*>(&r)       = *reinterpret_cast<const uint4*>(p);
  *(reinterpret_cast<uint4*>(&r) + 1) = *reinterpret_cast<const uint4*>(p + 16);
  return r;
}

__device__ __forceinline__ v8f wmma_bf16(v16bf a, v16bf b, v8f c) {
  return __builtin_amdgcn_wmma_f32_16x16x32_bf16(false, a, false, b,
                                                 (short)0, c, false, false);
}

// Scheduling-shape directives (soft): DS-read group then WMMA group, so the
// ds_loads clause up ahead of the matrix ops and waits retire progressively.
#define SCHED_DS_THEN_WMMA(nds, nw)                     \
  __builtin_amdgcn_sched_group_barrier(0x100, nds, 0);  \
  __builtin_amdgcn_sched_group_barrier(0x008, nw, 0)
#define SCHED_DS_VMEM_WMMA(nds, nvm, nw)                \
  __builtin_amdgcn_sched_group_barrier(0x100, nds, 0);  \
  __builtin_amdgcn_sched_group_barrier(0x020, nvm, 0);  \
  __builtin_amdgcn_sched_group_barrier(0x008, nw, 0)

// ---------------------------------------------------- weight preparation ---

__global__ void k_f32_to_bf16(const float* __restrict__ s, bf16* __restrict__ d, int n) {
  int i = blockIdx.x * blockDim.x + threadIdx.x;
  if (i < n) d[i] = (bf16)s[i];
}

// src (R,C) f32, batched over blockIdx.z -> dst (C,R) bf16 (transposed)
__global__ __launch_bounds__(256) void k_transpose_bf16(
    const float* __restrict__ src, bf16* __restrict__ dst, int R, int C) {
  __shared__ float tile[32][33];
  size_t bofs = (size_t)blockIdx.z * R * C;
  src += bofs; dst += bofs;
  int c0 = blockIdx.x * 32, r0 = blockIdx.y * 32;
  int tx = threadIdx.x & 31, ty = threadIdx.x >> 5;   // 32 x 8
#pragma unroll
  for (int i = 0; i < 4; ++i)
    tile[ty + i * 8][tx] = src[(size_t)(r0 + ty + i * 8) * C + c0 + tx];
  __syncthreads();
#pragma unroll
  for (int i = 0; i < 4; ++i)
    dst[(size_t)(c0 + ty + i * 8) * R + r0 + tx] = (bf16)tile[tx][ty + i * 8];
}

// ------------------------------------------------------------- LN + RoPE ---

__global__ __launch_bounds__(256) void k_ln_rope(
    const float* __restrict__ q, const float* __restrict__ k, const float* __restrict__ v,
    const float* __restrict__ s1, const float* __restrict__ b1,
    bf16* __restrict__ qn, bf16* __restrict__ kn, bf16* __restrict__ vn) {
  int t = blockIdx.x;
  int which = blockIdx.y;
  const float* src = (which == 0) ? q : (which == 1) ? k : v;
  bf16* dst        = (which == 0) ? qn : (which == 1) ? kn : vn;
  int tid = threadIdx.x;
  float x0 = src[(size_t)t * E_ + tid];
  float x1 = src[(size_t)t * E_ + tid + 256];
  float x2 = src[(size_t)t * E_ + tid + 512];
  float x3 = src[(size_t)t * E_ + tid + 768];
  __shared__ float red[256];
  red[tid] = x0 + x1 + x2 + x3;
  __syncthreads();
  for (int o = 128; o > 0; o >>= 1) { if (tid < o) red[tid] += red[tid + o]; __syncthreads(); }
  float mean = red[0] * (1.0f / E_);
  __syncthreads();
  float d0 = x0 - mean, d1 = x1 - mean, d2 = x2 - mean, d3 = x3 - mean;
  red[tid] = d0 * d0 + d1 * d1 + d2 * d2 + d3 * d3;
  __syncthreads();
  for (int o = 128; o > 0; o >>= 1) { if (tid < o) red[tid] += red[tid + o]; __syncthreads(); }
  float inv = rsqrtf(red[0] * (1.0f / E_) + 1e-5f);
  float n0 = d0 * inv * s1[tid]       + b1[tid];
  float n1 = d1 * inv * s1[tid + 256] + b1[tid + 256];
  float n2 = d2 * inv * s1[tid + 512] + b1[tid + 512];
  float n3 = d3 * inv * s1[tid + 768] + b1[tid + 768];
  float spos = (float)(t / B_);
  // rotary pairs (j, j+512), base 10000, half-split
  float f0 = spos * __expf(-9.210340372f * (float)tid * (1.0f / 512.0f));
  float c0, sn0; __sincosf(f0, &sn0, &c0);
  dst[(size_t)t * E_ + tid]       = (bf16)(n0 * c0 - n2 * sn0);
  dst[(size_t)t * E_ + tid + 512] = (bf16)(n0 * sn0 + n2 * c0);
  float f1 = spos * __expf(-9.210340372f * (float)(tid + 256) * (1.0f / 512.0f));
  float c1, sn1; __sincosf(f1, &sn1, &c1);
  dst[(size_t)t * E_ + tid + 256] = (bf16)(n1 * c1 - n3 * sn1);
  dst[(size_t)t * E_ + tid + 768] = (bf16)(n1 * sn1 + n3 * c1);
}

// -------------------------------------------------------- generic WMMA GEMM ---
// C[M x Nout] = A(bf16, lda) @ Bt^T + bias.  Bt stored as [n][k] (ldb).
// mode 0: bf16 out;  mode 1: f32 out with residual add.
// K-tile 64, double-buffered async global->LDS DMA: tile i+1's DMA flies
// under tile i's WMMAs (s_wait_asynccnt 0x6 keeps one tile outstanding).
__global__ __launch_bounds__(256) void k_gemm(
    const bf16* __restrict__ A, int lda,
    const bf16* __restrict__ Bt, int ldb,
    const float* __restrict__ bias, int K, int mode,
    bf16* __restrict__ Cb, float* __restrict__ Cf,
    const float* __restrict__ res, int ldc) {
  __shared__ __attribute__((aligned(16))) bf16 As[2][128 * 72];
  __shared__ __attribute__((aligned(16))) bf16 Bs[2][64 * 72];
  int tid = threadIdx.x, wave = tid >> 5;
  int lane = tid & 31;
  int row0 = blockIdx.x * 128, n0 = blockIdx.y * 64;

  // per-thread staging coordinates
  int ra = tid >> 1, ha = (tid & 1) * 32;          // A: 128 rows x 64 k, 64B run
  int rb = tid >> 2, hb = (tid & 3) * 16;          // B: 64 rows x 64 k, 32B run
  const bf16* gA = A  + (size_t)(row0 + ra) * lda + ha;
  const bf16* gB = Bt + (size_t)(n0   + rb) * ldb + hb;
  unsigned int lA[2] = { (unsigned int)(uintptr_t)(&As[0][ra * 72 + ha]),
                         (unsigned int)(uintptr_t)(&As[1][ra * 72 + ha]) };
  unsigned int lB[2] = { (unsigned int)(uintptr_t)(&Bs[0][rb * 72 + hb]),
                         (unsigned int)(uintptr_t)(&Bs[1][rb * 72 + hb]) };

  auto issue_tile = [&](int buf, int kk) {
    async_load16(lA[buf],      gA + kk);
    async_load16(lA[buf] + 16, gA + kk + 8);
    async_load16(lA[buf] + 32, gA + kk + 16);
    async_load16(lA[buf] + 48, gA + kk + 24);
    async_load16(lB[buf],      gB + kk);
    async_load16(lB[buf] + 16, gB + kk + 8);
  };

  v8f acc[4];
#pragma unroll
  for (int c = 0; c < 4; ++c) acc[c] = vzero8();

  int NT = K >> 6;                 // number of 64-wide K-tiles
  issue_tile(0, 0);
  if (NT > 1) issue_tile(1, 64);

  for (int i = 0; i < NT; ++i) {
    if (i + 1 < NT) wait_async6(); else wait_async0();
    __syncthreads();
    const bf16* as = As[i & 1];
    const bf16* bs = Bs[i & 1];
#pragma unroll
    for (int sub = 0; sub < 64; sub += 32) {
      v16bf a  = load_a_frag(as + wave * 16 * 72 + sub, 72);
      v16bf b0 = load_b_frag(bs + 0 * 16 * 72 + sub, 72);
      v16bf b1 = load_b_frag(bs + 1 * 16 * 72 + sub, 72);
      v16bf b2 = load_b_frag(bs + 2 * 16 * 72 + sub, 72);
      v16bf b3 = load_b_frag(bs + 3 * 16 * 72 + sub, 72);
      acc[0] = wmma_bf16(a, b0, acc[0]);
      acc[1] = wmma_bf16(a, b1, acc[1]);
      acc[2] = wmma_bf16(a, b2, acc[2]);
      acc[3] = wmma_bf16(a, b3, acc[3]);
      SCHED_DS_THEN_WMMA(10, 4);   // 10 ds_load_b128 then 4 v_wmma
    }
    __syncthreads();
    int kn = (i + 2) << 6;
    if (kn < K) issue_tile(i & 1, kn);
  }

  int hi = lane >> 4, cl = lane & 15;
#pragma unroll
  for (int c = 0; c < 4; ++c) {
#pragma unroll
    for (int r = 0; r < 8; ++r) {
      int row = row0 + wave * 16 + r + 8 * hi;
      int col = n0 + c * 16 + cl;
      float val = acc[c][r] + bias[col];
      if (mode == 0) Cb[(size_t)row * ldc + col] = (bf16)val;
      else           Cf[(size_t)row * ldc + col] = val + res[(size_t)row * ldc + col];
    }
  }
}

// ------------------------------------------------------- flash attention ---
// One wave per (16-row q-tile, b, h); online softmax; ctx written bf16.
__global__ __launch_bounds__(32) void k_attn(
    const bf16* __restrict__ qp, const bf16* __restrict__ kp,
    const bf16* __restrict__ vp, bf16* __restrict__ ctx) {
  __shared__ __attribute__((aligned(16))) bf16 Pb[16 * 40];
  __shared__ __attribute__((aligned(16))) bf16 Vt[64 * 40];
  int lane = threadIdx.x;
  int qt = blockIdx.x;
  int b  = blockIdx.y >> 4;
  int h  = blockIdx.y & 15;
  const int ld = B_ * E_;  // token row stride along seq
  const bf16* qbase = qp + ((size_t)(qt * 16) * B_ + b) * E_ + h * HD_;
  v16bf aq0 = load_a_frag(qbase, ld);
  v16bf aq1 = load_a_frag(qbase + 32, ld);
  v8f o0 = vzero8(), o1 = vzero8(), o2 = vzero8(), o3 = vzero8();
  float m_[8], l_[8];
#pragma unroll
  for (int r = 0; r < 8; ++r) { m_[r] = -1e30f; l_[r] = 0.0f; }
  int hi = lane >> 4, cl = lane & 15;

  for (int kc = 0; kc < S_ / 32; ++kc) {
    int ks = kc * 32;
    const bf16* kb0 = kp + ((size_t)ks * B_ + b) * E_ + h * HD_;
    const bf16* kb1 = kb0 + (size_t)16 * ld;
    // hoist the 4 K-fragments, then 4 WMMAs
    v16bf bk00 = load_b_frag(kb0,      ld);
    v16bf bk01 = load_b_frag(kb0 + 32, ld);
    v16bf bk10 = load_b_frag(kb1,      ld);
    v16bf bk11 = load_b_frag(kb1 + 32, ld);
    v8f s0 = vzero8(), s1 = vzero8();
    s0 = wmma_bf16(aq0, bk00, s0);
    s0 = wmma_bf16(aq1, bk01, s0);
    s1 = wmma_bf16(aq0, bk10, s1);
    s1 = wmma_bf16(aq1, bk11, s1);

#pragma unroll
    for (int r = 0; r < 8; ++r) {
      float v0 = s0[r] * 0.125f, v1 = s1[r] * 0.125f;   // 1/sqrt(64)
      float mx = fmaxf(v0, v1);
      mx = fmaxf(mx, __shfl_xor(mx, 1, 32));
      mx = fmaxf(mx, __shfl_xor(mx, 2, 32));
      mx = fmaxf(mx, __shfl_xor(mx, 4, 32));
      mx = fmaxf(mx, __shfl_xor(mx, 8, 32));
      float mnew = fmaxf(m_[r], mx);
      float corr = __expf(m_[r] - mnew);
      float p0 = __expf(v0 - mnew), p1 = __expf(v1 - mnew);
      float rs = p0 + p1;
      rs += __shfl_xor(rs, 1, 32);
      rs += __shfl_xor(rs, 2, 32);
      rs += __shfl_xor(rs, 4, 32);
      rs += __shfl_xor(rs, 8, 32);
      l_[r] = l_[r] * corr + rs;
      m_[r] = mnew;
      o0[r] *= corr; o1[r] *= corr; o2[r] *= corr; o3[r] *= corr;
      Pb[(r + 8 * hi) * 40 + cl]      = (bf16)p0;
      Pb[(r + 8 * hi) * 40 + 16 + cl] = (bf16)p1;
    }

    // stage V chunk transposed: Vt[dim][key]
    const bf16* vb = vp + ((size_t)ks * B_ + b) * E_ + h * HD_;
#pragma unroll
    for (int n = 0; n < HD_; ++n) Vt[n * 40 + lane] = vb[(size_t)lane * ld + n];
    __syncthreads();
    v16bf ap = load_a_frag(Pb, 40);
    v16bf bv0 = load_b_frag(Vt + 0 * 16 * 40, 40);
    v16bf bv1 = load_b_frag(Vt + 1 * 16 * 40, 40);
    v16bf bv2 = load_b_frag(Vt + 2 * 16 * 40, 40);
    v16bf bv3 = load_b_frag(Vt + 3 * 16 * 40, 40);
    o0 = wmma_bf16(ap, bv0, o0);
    o1 = wmma_bf16(ap, bv1, o1);
    o2 = wmma_bf16(ap, bv2, o2);
    o3 = wmma_bf16(ap, bv3, o3);
    __syncthreads();
  }

#pragma unroll
  for (int r = 0; r < 8; ++r) {
    float inv = 1.0f / l_[r];
    size_t tok = (size_t)(qt * 16 + r + 8 * hi) * B_ + b;
    bf16* cb = ctx + tok * E_ + h * HD_;
    cb[0 * 16 + cl] = (bf16)(o0[r] * inv);
    cb[1 * 16 + cl] = (bf16)(o1[r] * inv);
    cb[2 * 16 + cl] = (bf16)(o2[r] * inv);
    cb[3 * 16 + cl] = (bf16)(o3[r] * inv);
  }
}

// ----------------------------------------------- LN2 (+ seed output), MoE ---

__global__ __launch_bounds__(256) void k_ln2(
    const float* __restrict__ x, const float* __restrict__ s2, const float* __restrict__ b2,
    bf16* __restrict__ xn, float* __restrict__ out) {
  int t = blockIdx.x, tid = threadIdx.x;
  float xv[4];
#pragma unroll
  for (int i = 0; i < 4; ++i) xv[i] = x[(size_t)t * E_ + tid + 256 * i];
  __shared__ float red[256];
  red[tid] = xv[0] + xv[1] + xv[2] + xv[3];
  __syncthreads();
  for (int o = 128; o > 0; o >>= 1) { if (tid < o) red[tid] += red[tid + o]; __syncthreads(); }
  float mean = red[0] * (1.0f / E_);
  __syncthreads();
  float ss = 0.0f;
#pragma unroll
  for (int i = 0; i < 4; ++i) { float d = xv[i] - mean; ss += d * d; }
  red[tid] = ss;
  __syncthreads();
  for (int o = 128; o > 0; o >>= 1) { if (tid < o) red[tid] += red[tid + o]; __syncthreads(); }
  float inv = rsqrtf(red[0] * (1.0f / E_) + 1e-5f);
#pragma unroll
  for (int i = 0; i < 4; ++i) {
    int e = tid + 256 * i;
    xn[(size_t)t * E_ + e]  = (bf16)((xv[i] - mean) * inv * s2[e] + b2[e]);
    out[(size_t)t * E_ + e] = xv[i];   // residual seed; experts add on top
  }
}

__global__ __launch_bounds__(256) void k_router(
    const bf16* __restrict__ xn, const float* __restrict__ rw, int* __restrict__ eid) {
  int t = blockIdx.x, tid = threadIdx.x;
  int wv = tid >> 5, lane = tid & 31;   // wave wv computes logit for expert wv
  float acc = 0.0f;
  for (int k = lane; k < E_; k += 32) acc += (float)xn[(size_t)t * E_ + k] * rw[(size_t)k * NE_ + wv];
  for (int o = 16; o > 0; o >>= 1) acc += __shfl_xor(acc, o, 32);
  __shared__ float lg[NE_];
  if (lane == 0) lg[wv] = acc;
  __syncthreads();
  if (tid == 0) {
    int best = 0; float bv = lg[0];
    for (int e = 1; e < NE_; ++e) if (lg[e] > bv) { bv = lg[e]; best = e; }
    eid[t] = best;
  }
}

__global__ void k_capacity(const int* __restrict__ eid,
                           int* __restrict__ idx, int* __restrict__ counts) {
  int e = threadIdx.x;
  if (e >= NE_) return;
  int c = 0;
  for (int t = 0; t < N_; ++t)
    if (eid[t] == e) { if (c < CAP_) idx[e * CAP_ + c] = t; ++c; }
  counts[e] = (c > CAP_) ? CAP_ : c;
}

// Per (expert, 16-row gathered tile): FFN1 -> GELU (LDS) -> FFN2 -> scatter-add.
// Token gather uses async global->LDS DMA; 4 output tiles per inner K-loop so
// one A-fragment ds_load feeds 4 WMMAs; sched_group_barrier pins the
// DS -> VMEM -> WMMA pipeline shape.
__global__ __launch_bounds__(256) void k_moe(
    const bf16* __restrict__ xn, const bf16* __restrict__ wiT, const bf16* __restrict__ woT,
    const int* __restrict__ idx, const int* __restrict__ counts, float* __restrict__ out) {
  extern __shared__ __attribute__((aligned(16))) char smem[];
  bf16* As = reinterpret_cast<bf16*>(smem);                       // 16 x 1032
  bf16* Hs = reinterpret_cast<bf16*>(smem + 16 * 1032 * 2);      // 16 x 4104
  int e = blockIdx.y, tile = blockIdx.x;
  int cnt = counts[e];
  int nrows = cnt - tile * 16;
  if (nrows <= 0) return;
  if (nrows > 16) nrows = 16;
  const int* rows = idx + e * CAP_ + tile * 16;
  int tid = threadIdx.x, wave = tid >> 5, lane = tid & 31;
  int hi = lane >> 4, cl = lane & 15;

  { // gather A: 16 rows x 1024, async DMA (8 x 16B per thread)
    int r = tid >> 4, chunk = tid & 15;
    int tok = rows[r < nrows ? r : 0];
    const bf16* g = xn + (size_t)tok * E_ + chunk * 64;
    unsigned int l = (unsigned int)(uintptr_t)(As + r * 1032 + chunk * 64);
#pragma unroll
    for (int i = 0; i < 8; ++i) async_load16(l + 16 * i, g + 8 * i);
    wait_async0();
  }
  __syncthreads();

  const bf16* wi_e = wiT + (size_t)e * F_ * E_;   // (F, E): row f, k contiguous
  const bf16* wo_e = woT + (size_t)e * E_ * F_;   // (E, F): row n, k contiguous

  // phase 1: H = gelu(A @ wi[e]) : wave covers 512 F-columns, 4 tiles at a time
  for (int ft = 0; ft < 8; ++ft) {
    int n0 = wave * 512 + ft * 64;
    v8f a0 = vzero8(), a1 = vzero8(), a2 = vzero8(), a3 = vzero8();
    for (int kk = 0; kk < 32; ++kk) {
      v16bf a = load_a_frag(As + kk * 32, 1032);
      const bf16* wp = wi_e + (size_t)n0 * E_ + kk * 32;
      v16bf b0 = load_b_frag(wp,                   E_);
      v16bf b1 = load_b_frag(wp + (size_t)16 * E_, E_);
      v16bf b2 = load_b_frag(wp + (size_t)32 * E_, E_);
      v16bf b3 = load_b_frag(wp + (size_t)48 * E_, E_);
      a0 = wmma_bf16(a, b0, a0);
      a1 = wmma_bf16(a, b1, a1);
      a2 = wmma_bf16(a, b2, a2);
      a3 = wmma_bf16(a, b3, a3);
      SCHED_DS_VMEM_WMMA(2, 8, 4);   // a-frag ds, 4 b-frags vmem, 4 wmma
    }
#pragma unroll
    for (int r = 0; r < 8; ++r) {
      int rr = r + 8 * hi;
      float x0 = a0[r], x1 = a1[r], x2 = a2[r], x3 = a3[r];
      Hs[rr * 4104 + n0 +  0 + cl] = (bf16)(0.5f * x0 * (1.0f + erff(x0 * 0.70710678f)));
      Hs[rr * 4104 + n0 + 16 + cl] = (bf16)(0.5f * x1 * (1.0f + erff(x1 * 0.70710678f)));
      Hs[rr * 4104 + n0 + 32 + cl] = (bf16)(0.5f * x2 * (1.0f + erff(x2 * 0.70710678f)));
      Hs[rr * 4104 + n0 + 48 + cl] = (bf16)(0.5f * x3 * (1.0f + erff(x3 * 0.70710678f)));
    }
  }
  __syncthreads();

  // phase 2: out += H @ wo[e] : wave covers 128 out-columns, 4 tiles at a time
  for (int ot = 0; ot < 2; ++ot) {
    int n0 = wave * 128 + ot * 64;
    v8f a0 = vzero8(), a1 = vzero8(), a2 = vzero8(), a3 = vzero8();
    for (int kk = 0; kk < 128; ++kk) {
      v16bf a = load_a_frag(Hs + kk * 32, 4104);
      const bf16* wp = wo_e + (size_t)n0 * F_ + kk * 32;
      v16bf b0 = load_b_frag(wp,                   F_);
      v16bf b1 = load_b_frag(wp + (size_t)16 * F_, F_);
      v16bf b2 = load_b_frag(wp + (size_t)32 * F_, F_);
      v16bf b3 = load_b_frag(wp + (size_t)48 * F_, F_);
      a0 = wmma_bf16(a, b0, a0);
      a1 = wmma_bf16(a, b1, a1);
      a2 = wmma_bf16(a, b2, a2);
      a3 = wmma_bf16(a, b3, a3);
      SCHED_DS_VMEM_WMMA(2, 8, 4);
    }
#pragma unroll
    for (int r = 0; r < 8; ++r) {
      int rr = r + 8 * hi;
      if (rr < nrows) {
        int tok = rows[rr];
        float* op = out + (size_t)tok * E_ + n0 + cl;
        op[ 0] += a0[r];
        op[16] += a1[r];
        op[32] += a2[r];
        op[48] += a3[r];
      }
    }
  }
}

// ------------------------------------------------------------------ host ---

extern "C" void kernel_launch(void* const* d_in, const int* in_sizes, int n_in,
                              void* d_out, int out_size, void* d_ws, size_t ws_size,
                              hipStream_t stream) {
  (void)in_sizes; (void)n_in; (void)out_size; (void)ws_size;
  const float* q     = (const float*)d_in[0];
  const float* k     = (const float*)d_in[1];
  const float* v     = (const float*)d_in[2];
  const float* w_in  = (const float*)d_in[3];
  const float* b_in  = (const float*)d_in[4];
  const float* w_out = (const float*)d_in[5];
  const float* b_out = (const float*)d_in[6];
  const float* n1s   = (const float*)d_in[7];
  const float* n1b   = (const float*)d_in[8];
  const float* n2s   = (const float*)d_in[9];
  const float* n2b   = (const float*)d_in[10];
  const float* rw    = (const float*)d_in[11];
  const float* wi    = (const float*)d_in[12];
  const float* wo    = (const float*)d_in[13];
  float* out = (float*)d_out;

  char* ws = (char*)d_ws;
  size_t off = 0;
  auto alloc = [&](size_t bytes) {
    char* p = ws + off;
    off += (bytes + 255) & ~(size_t)255;
    return p;
  };
  const size_t NB = (size_t)N_ * E_ * sizeof(bf16);   // 16 MB
  bf16* qn  = (bf16*)alloc(NB);
  bf16* kn  = (bf16*)alloc(NB);
  bf16* vn  = (bf16*)alloc(NB);
  bf16* qp  = (bf16*)alloc(NB);
  bf16* kp  = (bf16*)alloc(NB);
  bf16* vp  = (bf16*)alloc(NB);
  bf16* ctx = (bf16*)alloc(NB);
  bf16* xnb = (bf16*)alloc(NB);
  float* xres = (float*)alloc((size_t)N_ * E_ * sizeof(float));
  bf16* w_in_bf  = (bf16*)alloc((size_t)3 * E_ * E_ * sizeof(bf16));
  bf16* w_out_bf = (bf16*)alloc((size_t)E_ * E_ * sizeof(bf16));
  bf16* wiT = (bf16*)alloc((size_t)NE_ * F_ * E_ * sizeof(bf16));
  bf16* woT = (bf16*)alloc((size_t)NE_ * E_ * F_ * sizeof(bf16));
  int* eid    = (int*)alloc((size_t)N_ * sizeof(int));
  int* idx    = (int*)alloc((size_t)NE_ * CAP_ * sizeof(int));
  int* counts = (int*)alloc((size_t)NE_ * sizeof(int));

  // 1) weight conversion (bf16, B-fragment-friendly transposed layouts)
  { int n = 3 * E_ * E_; k_f32_to_bf16<<<(n + 255) / 256, 256, 0, stream>>>(w_in, w_in_bf, n); }
  { int n = E_ * E_;     k_f32_to_bf16<<<(n + 255) / 256, 256, 0, stream>>>(w_out, w_out_bf, n); }
  k_transpose_bf16<<<dim3(F_ / 32, E_ / 32, NE_), 256, 0, stream>>>(wi, wiT, E_, F_);
  k_transpose_bf16<<<dim3(E_ / 32, F_ / 32, NE_), 256, 0, stream>>>(wo, woT, F_, E_);

  // 2) LN + RoPE for q/k/v
  k_ln_rope<<<dim3(N_, 3), 256, 0, stream>>>(q, k, v, n1s, n1b, qn, kn, vn);

  // 3) in_proj (three 8192x1024x1024 GEMMs)
  dim3 gg(N_ / 128, E_ / 64);
  k_gemm<<<gg, 256, 0, stream>>>(qn, E_, w_in_bf,                       E_, b_in,          E_, 0, qp, nullptr, nullptr, E_);
  k_gemm<<<gg, 256, 0, stream>>>(kn, E_, w_in_bf + (size_t)E_ * E_,     E_, b_in + E_,     E_, 0, kp, nullptr, nullptr, E_);
  k_gemm<<<gg, 256, 0, stream>>>(vn, E_, w_in_bf + (size_t)2 * E_ * E_, E_, b_in + 2 * E_, E_, 0, vp, nullptr, nullptr, E_);

  // 4) flash attention
  k_attn<<<dim3(S_ / 16, B_ * H_), 32, 0, stream>>>(qp, kp, vp, ctx);

  // 5) out_proj + residual(query) -> x
  k_gemm<<<gg, 256, 0, stream>>>(ctx, E_, w_out_bf, E_, b_out, E_, 1, nullptr, xres, q, E_);

  // 6) LN2 (writes xn, seeds d_out with residual)
  k_ln2<<<N_, 256, 0, stream>>>(xres, n2s, n2b, xnb, out);

  // 7) router + capacity
  k_router<<<N_, 256, 0, stream>>>(xnb, rw, eid);
  k_capacity<<<1, 32, 0, stream>>>(eid, idx, counts);

  // 8) expert FFNs on gathered tokens
  size_t moe_smem = (size_t)(16 * 1032 + 16 * 4104) * sizeof(bf16);  // ~160.5 KB
  k_moe<<<dim3(CAP_ / 16, NE_), 256, moe_smem, stream>>>(xnb, wiT, woT, idx, counts, out);
}
